// ElmanLeakyMamba2Delta_86466281603868
// MI455X (gfx1250) — matrix operand order repaired and checked by reading, not compiled, not run on hardware
//
#include <hip/hip_runtime.h>
#include <math.h>

// Problem sizes (fixed by the reference)
#define T_STEPS 2048
#define BATCH   8
#define D_INP   1024
#define H_DIM   1024
#define M_ROWS  (T_STEPS * BATCH)          // 16384
#define NWG_SCAN 8
#define COLS_PER_WG (H_DIM / NWG_SCAN)     // 128

// Workspace layout (bytes)
#define OFF_XW   ((size_t)0)                       // f32 [M, H]          64 MB
#define OFF_DEC  ((size_t)67108864)                // f32 [M, H]          64 MB
#define OFF_XBF  ((size_t)134217728)               // bf16 [M, D]         32 MB
#define OFF_WBF  ((size_t)167772160)               // bf16 [2H, D]         4 MB
#define OFF_RBF  ((size_t)171966464)               // bf16 [H, H]          2 MB
#define OFF_HBF  ((size_t)174063616)               // bf16 [2][16][H]     64 KB (ping-pong)
#define OFF_HF   ((size_t)174129152)               // f32 [16][H]         64 KB
#define OFF_BAR  ((size_t)174194688)               // 2 x u32 barrier

typedef __bf16 v8bf  __attribute__((ext_vector_type(8)));
typedef __bf16 v16bf __attribute__((ext_vector_type(16)));
typedef float  v8f   __attribute__((ext_vector_type(8)));
typedef unsigned int v4u __attribute__((ext_vector_type(4)));
typedef int v8i __attribute__((ext_vector_type(8)));
typedef int v4i __attribute__((ext_vector_type(4)));

#define HAVE_TDM (__has_builtin(__builtin_amdgcn_tensor_load_to_lds) && \
                  __has_builtin(__builtin_amdgcn_s_wait_tensorcnt))

__device__ __forceinline__ unsigned short f2bf(float f) {
  union { float f; unsigned u; } c; c.f = f;
  unsigned u = c.u;
  unsigned r = u + 0x7FFFu + ((u >> 16) & 1u);       // round-to-nearest-even
  if ((u & 0x7F800000u) == 0x7F800000u) r = u;       // inf/nan passthrough
  return (unsigned short)(r >> 16);
}

// Build a v16bf WMMA operand from two contiguous 16B groups (halves 0..7, 8..15).
__device__ __forceinline__ v16bf ldpair(const unsigned short* p0, const unsigned short* p1) {
  v8bf a = *(const v8bf*)p0;
  v8bf b = *(const v8bf*)p1;
  return __builtin_shufflevector(a, b, 0,1,2,3,4,5,6,7,8,9,10,11,12,13,14,15);
}

#if HAVE_TDM
// Flat shared-aperture pointer -> 32-bit LDS byte address (low 32 bits of the pointer).
__device__ __forceinline__ unsigned lds_addr_of(const void* p) {
  return (unsigned)(unsigned long long)(size_t)p;
}

// Issue a 2D (or 1D when tile_d1==0) bf16 TDM load: global -> LDS.
// D# layout per CDNA5 ISA; 6-arg builtin form
// (uint32x4 g0, int32x8 g1, int32x4 g2, int32x4 g3, int32x8, i32 cpol).
__device__ __forceinline__ void tdm_load_2d_bf16(unsigned lds_off, const void* gptr,
                                                 unsigned tensor_d0, unsigned tensor_d1,
                                                 unsigned tile_d0, unsigned tile_d1,
                                                 unsigned stride0_elems) {
  unsigned long long ga = (unsigned long long)(size_t)gptr;
  v4u g0;
  g0.x = 1u;                                           // count=1 (valid user D#)
  g0.y = lds_off;                                      // lds_addr [63:32]
  g0.z = (unsigned)(ga & 0xFFFFFFFFu);                 // global_addr [95:64]
  g0.w = (unsigned)((ga >> 32) & 0x01FFFFFFu)          // global_addr [120:96]
       | (2u << 30);                                   // type=2 ("image")
  v8i g1;
  g1[0] = (int)(1u << 16);                             // wg_mask=0, data_size=1 (2B)
  g1[1] = (int)((tensor_d0 & 0xFFFFu) << 16);          // tensor_dim0[15:0] @ bits 63:48
  g1[2] = (int)(((tensor_d0 >> 16) & 0xFFFFu)          // tensor_dim0[31:16]
       | ((tensor_d1 & 0xFFFFu) << 16));               // tensor_dim1[15:0]
  g1[3] = (int)(((tensor_d1 >> 16) & 0xFFFFu)          // tensor_dim1[31:16]
       | ((tile_d0 & 0xFFFFu) << 16));                 // tile_dim0
  g1[4] = (int)(tile_d1 & 0xFFFFu);                    // tile_dim1 (0 => 1D), tile_dim2=0
  g1[5] = (int)stride0_elems;                          // tensor_dim0_stride[31:0]
  g1[6] = 0;                                           // stride0 hi, dim1_stride lo
  g1[7] = 0;
  v4i z4 = {0, 0, 0, 0};
  v8i z8 = {0, 0, 0, 0, 0, 0, 0, 0};
  __builtin_amdgcn_tensor_load_to_lds(g0, g1, z4, z4, z8, 0);
}
#endif

// ---------------- pack kernels ----------------

__global__ __launch_bounds__(256) void pack_weights(
    const float* __restrict__ Wx, const float* __restrict__ Wd,
    const float* __restrict__ R,
    unsigned short* __restrict__ wbf, unsigned short* __restrict__ rbf) {
  size_t i = (size_t)blockIdx.x * blockDim.x + threadIdx.x;
  const size_t NW = (size_t)2 * H_DIM * D_INP;
  const size_t NR = (size_t)H_DIM * H_DIM;
  if (i < NW) {
    float v = (i < (size_t)H_DIM * D_INP) ? Wx[i] : Wd[i - (size_t)H_DIM * D_INP];
    wbf[i] = f2bf(v);
  } else if (i < NW + NR) {
    rbf[i - NW] = f2bf(R[i - NW]);
  }
}

__global__ __launch_bounds__(256) void pack_x(
    const float* __restrict__ x, unsigned short* __restrict__ xbf) {
  size_t i = (size_t)blockIdx.x * blockDim.x + threadIdx.x; // one float4 each
  float4 v = ((const float4*)x)[i];
  ushort4 o;
  o.x = f2bf(v.x); o.y = f2bf(v.y); o.z = f2bf(v.z); o.w = f2bf(v.w);
  ((ushort4*)xbf)[i] = o;
}

__global__ __launch_bounds__(256) void init_state(
    const float* __restrict__ h0,
    unsigned short* __restrict__ hbf, float* __restrict__ hf,
    unsigned* __restrict__ bar) {
  int i = blockIdx.x * blockDim.x + threadIdx.x;     // 16*H threads
  if (i < 16 * H_DIM) {
    int r = i / H_DIM, c = i % H_DIM;
    float v = (r < BATCH) ? h0[r * H_DIM + c] : 0.f;
    hf[i] = v;
    unsigned short b = f2bf(v);
    hbf[i] = b;
    hbf[16 * H_DIM + i] = b;                         // both ping-pong buffers
  }
  if (i < 2) bar[i] = 0u;
}

// ---------------- fused input-projection GEMM (bf16 WMMA) ----------------
// out[m, n] = sum_k x[m,k] * Wcat[n,k];   n<H -> xw+bias, n>=H -> decay

__global__ __launch_bounds__(256) void gemm_xw_decay(
    const unsigned short* __restrict__ xbf,          // [M, D] bf16
    const unsigned short* __restrict__ wbf,          // [2H, D] bf16
    const float* __restrict__ bias, const float* __restrict__ b_delta,
    float* __restrict__ xw, float* __restrict__ dec) {
  const int lane  = threadIdx.x & 31;
  const int wave  = threadIdx.x >> 5;
  const int mTile = blockIdx.x;                      // 0..1023
  const int nBase = (blockIdx.y * 8 + wave) * 64;    // 0..1984
  const int l15   = lane & 15;
  const int off8  = (lane < 16) ? 0 : 8;             // A: K 0-7/16-23 vs 8-15/24-31
  const int off16 = (lane < 16) ? 0 : 16;            // unused; B is loaded with off8, same K order as A

  const unsigned short* xrow = xbf + (size_t)(mTile * 16 + l15) * D_INP;
  const unsigned short* w0 = wbf + (size_t)(nBase +  0 + l15) * D_INP;
  const unsigned short* w1 = wbf + (size_t)(nBase + 16 + l15) * D_INP;
  const unsigned short* w2 = wbf + (size_t)(nBase + 32 + l15) * D_INP;
  const unsigned short* w3 = wbf + (size_t)(nBase + 48 + l15) * D_INP;

  v8f acc[4];
  acc[0] = (v8f)0.f; acc[1] = (v8f)0.f; acc[2] = (v8f)0.f; acc[3] = (v8f)0.f;

#pragma unroll 4
  for (int k0 = 0; k0 < D_INP; k0 += 32) {
    v16bf A  = ldpair(xrow + k0 + off8, xrow + k0 + 16 + off8);
    v16bf B0 = ldpair(w0 + k0 + off8, w0 + k0 + 16 + off8);
    acc[0] = __builtin_amdgcn_wmma_f32_16x16x32_bf16(false, A, false, B0, (short)0, acc[0], false, false);
    v16bf B1 = ldpair(w1 + k0 + off8, w1 + k0 + 16 + off8);
    acc[1] = __builtin_amdgcn_wmma_f32_16x16x32_bf16(false, A, false, B1, (short)0, acc[1], false, false);
    v16bf B2 = ldpair(w2 + k0 + off8, w2 + k0 + 16 + off8);
    acc[2] = __builtin_amdgcn_wmma_f32_16x16x32_bf16(false, A, false, B2, (short)0, acc[2], false, false);
    v16bf B3 = ldpair(w3 + k0 + off8, w3 + k0 + 16 + off8);
    acc[3] = __builtin_amdgcn_wmma_f32_16x16x32_bf16(false, A, false, B3, (short)0, acc[3], false, false);
  }

  const int mBase = mTile * 16 + ((lane < 16) ? 0 : 8);  // C/D layout: VGPR v -> M = v (+8 hi half)
#pragma unroll
  for (int nt = 0; nt < 4; ++nt) {
    const int col = nBase + nt * 16 + l15;
    if (col < H_DIM) {
      const float bv = bias[col];
#pragma unroll
      for (int v = 0; v < 8; ++v)
        xw[(size_t)(mBase + v) * H_DIM + col] = acc[nt][v] + bv;
    } else {
      const int c2 = col - H_DIM;
      const float bd = b_delta[c2];
#pragma unroll
      for (int v = 0; v < 8; ++v) {
        float d  = acc[nt][v] + bd;
        float sp = (d > 20.f) ? d : log1pf(__expf(d));   // softplus
        dec[(size_t)(mBase + v) * H_DIM + c2] = __expf(-sp);
      }
    }
  }
}

// ---------------- persistent recurrent scan (8 WGs, LDS-resident R slice) ----------------

__global__ __launch_bounds__(256) void scan_recurrent(
    const unsigned short* __restrict__ rbf,          // [H, H] bf16 (row j = R[j][:])
    const float* __restrict__ xw, const float* __restrict__ dec,
    unsigned short* __restrict__ hbf,                // [2][16][H] bf16 ping-pong
    float* __restrict__ hf,                          // [16][H] f32 state
    float* __restrict__ out,                         // [T, B, H] f32
    unsigned* __restrict__ bar) {
  extern __shared__ unsigned char smem[];
  unsigned short* r_lds = (unsigned short*)smem;                              // 128*1024 bf16 = 256 KB
  unsigned short* h_lds = (unsigned short*)(smem + (size_t)COLS_PER_WG * H_DIM * 2); // 16*1024 = 32 KB

  const int wgBase = blockIdx.x * COLS_PER_WG;
  const int lane = threadIdx.x & 31;
  const int wave = threadIdx.x >> 5;                 // 0..7, each owns 16 columns

  // Stage this WG's R slice (rows wgBase..wgBase+127) into LDS once.
#if HAVE_TDM
  if (wave == 0) {
    tdm_load_2d_bf16(lds_addr_of(r_lds), rbf + (size_t)wgBase * H_DIM,
                     /*tensor_d0=*/H_DIM, /*tensor_d1=*/COLS_PER_WG,
                     /*tile_d0=*/H_DIM, /*tile_d1=*/COLS_PER_WG,
                     /*stride0=*/H_DIM);
    __builtin_amdgcn_s_wait_tensorcnt(0);
  }
#else
  {
    const uint4* src = (const uint4*)(rbf + (size_t)wgBase * H_DIM);
    uint4* dst = (uint4*)r_lds;
    const int n16 = COLS_PER_WG * H_DIM / 8;
    for (int i = threadIdx.x; i < n16; i += blockDim.x) dst[i] = src[i];
  }
#endif

  const int l15  = lane & 15;
  const int off8  = (lane < 16) ? 0 : 8;
  const int off16 = (lane < 16) ? 0 : 16;
  const int colLocal = wave * 16 + l15;
  const int col = wgBase + colLocal;
  const unsigned short* rrow = r_lds + (size_t)colLocal * H_DIM;
  const unsigned short* hrow = h_lds + (size_t)l15 * H_DIM;

  for (int t = 0; t < T_STEPS; ++t) {
    // Stage current h (bf16, 16x1024 = 32 KB) into LDS via the TDM.
#if HAVE_TDM
    if (wave == 0) {
      tdm_load_2d_bf16(lds_addr_of(h_lds), hbf + (size_t)(t & 1) * 16 * H_DIM,
                       /*tensor_d0=*/16 * H_DIM, /*tensor_d1=*/1,
                       /*tile_d0=*/16 * H_DIM, /*tile_d1=*/0,   // 1D tile
                       /*stride0=*/16 * H_DIM);
      __builtin_amdgcn_s_wait_tensorcnt(0);
    }
#else
    {
      const uint4* s = (const uint4*)(hbf + (size_t)(t & 1) * 16 * H_DIM);
      uint4* d = (uint4*)h_lds;
      for (int i = threadIdx.x; i < 16 * H_DIM / 8; i += blockDim.x) d[i] = s[i];
    }
#endif
    __syncthreads();

    // Prefetch next step's xw/decay rows into cache while WMMAs run.
    if (t + 1 < T_STEPS && lane < 16) {
      const size_t nb = (size_t)(t + 1) * BATCH * H_DIM + col;
#pragma unroll
      for (int v = 0; v < 8; ++v) {
        __builtin_prefetch(xw + nb + (size_t)v * H_DIM, 0, 1);
        __builtin_prefetch(dec + nb + (size_t)v * H_DIM, 0, 1);
      }
    }

    // acc = h @ R_slice^T   (16x16 tile per wave, K = 1024, all operands from LDS)
    v8f acc = (v8f)0.f;
#pragma unroll 8
    for (int k0 = 0; k0 < H_DIM; k0 += 32) {
      v16bf A = ldpair(hrow + k0 + off8, hrow + k0 + 16 + off8);
      v16bf B = ldpair(rrow + k0 + off8, rrow + k0 + 16 + off8);
      acc = __builtin_amdgcn_wmma_f32_16x16x32_bf16(false, A, false, B, (short)0, acc, false, false);
    }

    unsigned short* hdst = hbf + (size_t)((t + 1) & 1) * 16 * H_DIM;
    if (lane < 16) {                                 // rows 0..7 = real batches
      const size_t tb = (size_t)t * BATCH * H_DIM;
#pragma unroll
      for (int v = 0; v < 8; ++v) {
        const size_t idx = tb + (size_t)v * H_DIM + col;
        const float xv = xw[idx];
        const float dv = dec[idx];
        const float hp = hf[v * H_DIM + col];
        const float cand = tanhf(acc[v] + xv);
        const float hn = dv * hp + (1.f - dv) * cand;
        hf[v * H_DIM + col] = hn;
        out[idx] = hn;
        hdst[v * H_DIM + col] = f2bf(hn);
      }
    } else {                                         // rows 8..15 stay zero-padded
#pragma unroll
      for (int v = 0; v < 8; ++v) hdst[(8 + v) * H_DIM + col] = 0;
    }

    // Device-scope sense-reversing grid barrier across the 8 persistent WGs.
    __syncthreads();
    if (threadIdx.x == 0) {
      __threadfence();
      unsigned g = __hip_atomic_load(&bar[1], __ATOMIC_RELAXED, __HIP_MEMORY_SCOPE_AGENT);
      unsigned a = __hip_atomic_fetch_add(&bar[0], 1u, __ATOMIC_ACQ_REL, __HIP_MEMORY_SCOPE_AGENT);
      if (a == NWG_SCAN - 1) {
        __hip_atomic_store(&bar[0], 0u, __ATOMIC_RELAXED, __HIP_MEMORY_SCOPE_AGENT);
        __hip_atomic_fetch_add(&bar[1], 1u, __ATOMIC_RELEASE, __HIP_MEMORY_SCOPE_AGENT);
      } else {
        while (__hip_atomic_load(&bar[1], __ATOMIC_ACQUIRE, __HIP_MEMORY_SCOPE_AGENT) == g)
          __builtin_amdgcn_s_sleep(1);
      }
      __threadfence();
    }
    __syncthreads();
  }
}

// ---------------- launcher ----------------

extern "C" void kernel_launch(void* const* d_in, const int* in_sizes, int n_in,
                              void* d_out, int out_size, void* d_ws, size_t ws_size,
                              hipStream_t stream) {
  const float* x    = (const float*)d_in[0];
  const float* h0   = (const float*)d_in[1];
  const float* Wx   = (const float*)d_in[2];
  const float* R    = (const float*)d_in[3];
  const float* bias = (const float*)d_in[4];
  const float* Wd   = (const float*)d_in[5];
  const float* bd   = (const float*)d_in[6];
  float* out = (float*)d_out;

  char* ws = (char*)d_ws;
  float*          xw  = (float*)(ws + OFF_XW);
  float*          dec = (float*)(ws + OFF_DEC);
  unsigned short* xbf = (unsigned short*)(ws + OFF_XBF);
  unsigned short* wbf = (unsigned short*)(ws + OFF_WBF);
  unsigned short* rbf = (unsigned short*)(ws + OFF_RBF);
  unsigned short* hbf = (unsigned short*)(ws + OFF_HBF);
  float*          hf  = (float*)(ws + OFF_HF);
  unsigned*       bar = (unsigned*)(ws + OFF_BAR);

  // 1) bf16 packing of weights and x
  pack_weights<<<(2 * H_DIM * D_INP + H_DIM * H_DIM + 255) / 256, 256, 0, stream>>>(Wx, Wd, R, wbf, rbf);
  pack_x<<<(M_ROWS * D_INP / 4) / 256, 256, 0, stream>>>(x, xbf);

  // 2) state + barrier init (idempotent per call)
  init_state<<<(16 * H_DIM + 255) / 256, 256, 0, stream>>>(h0, hbf, hf, bar);

  // 3) fused xw / decay projection GEMM (bf16 WMMA, f32 accumulate)
  gemm_xw_decay<<<dim3(M_ROWS / 16, 4), 256, 0, stream>>>(xbf, wbf, bias, bd, xw, dec);

  // 4) persistent recurrent scan: 8 WGs, 288 KB dynamic LDS each (R slice + h tile)
  const size_t smem = (size_t)(COLS_PER_WG * H_DIM + 16 * H_DIM) * sizeof(unsigned short); // 294912 B
  scan_recurrent<<<NWG_SCAN, 256, smem, stream>>>(rbf, xw, dec, hbf, hf, out, bar);
}
